// SparseMHA_23785528886210
// MI455X (gfx1250) — matrix operand order repaired and checked by reading, not compiled, not run on hardware
//
#include <hip/hip_runtime.h>
#include <hip/hip_bf16.h>

#define HID 128
#define H 8
#define DH 16

typedef __attribute__((ext_vector_type(16))) __bf16 v16bf;
typedef __attribute__((ext_vector_type(8)))  float  v8f;

// ---------------------------------------------------------------------------
// Kernel 0: pre-pack Wq/Wk/Wv into WMMA B-operand lane layout, bf16 hi/lo.
// B = W^T, so B column j (K sequential) == row j of W (contiguous).
// Storage: flat index ((mat*8 + c)*4 + kb)*32 + lane -> 16 bf16 (32 bytes).
//   lane<16 : col j = 16c+lane,    K = 32kb   .. 32kb+15
//   lane>=16: col j = 16c+lane-16, K = 32kb+16.. 32kb+31
// ---------------------------------------------------------------------------
__global__ void pack_w_kernel(const float* __restrict__ Wq,
                              const float* __restrict__ Wk,
                              const float* __restrict__ Wv,
                              __bf16* __restrict__ bhi, __bf16* __restrict__ blo)
{
    const int idx = blockIdx.x * blockDim.x + threadIdx.x;   // 0 .. 3071
    if (idx >= 3 * 8 * 4 * 32) return;
    const int lane = idx & 31;
    const int kb   = (idx >> 5) & 3;
    const int c    = (idx >> 7) & 7;
    const int mat  = idx >> 10;
    const float* W = (mat == 0) ? Wq : ((mat == 1) ? Wk : Wv);
    const int j    = (c << 4) + (lane & 15);
    const int half = lane >> 4;
    const float* wp = W + (size_t)j * HID + kb * 32 + half * 16;
    __bf16* ph = bhi + (size_t)idx * 16;
    __bf16* pl = blo + (size_t)idx * 16;
#pragma unroll
    for (int t = 0; t < 16; ++t) {
        float f  = wp[t];
        __bf16 x = (__bf16)f;
        ph[t] = x;
        pl[t] = (__bf16)(f - (float)x);
    }
}

// ---------------------------------------------------------------------------
// Kernel 1: fused QKV projection, y = h @ W.T + b, written HEAD-MAJOR:
//   out[n, hh*16 + d] = proj[n, d*8 + hh]   (q additionally scaled by DH^-0.5)
// One wave = one 16-row strip; A split to bf16 hi/lo once, B operands are
// pre-packed -> inner loop is 2x 32-byte loads + 3x v_wmma per 32-wide K chunk.
// ---------------------------------------------------------------------------
__global__ __launch_bounds__(256) void qkv_proj_wmma(
    const float* __restrict__ h,
    const __bf16* __restrict__ wbhi, const __bf16* __restrict__ wblo,
    const float* __restrict__ bq, const float* __restrict__ bk,
    const float* __restrict__ bv,
    float* __restrict__ qh, float* __restrict__ kh, float* __restrict__ vh,
    int N)
{
    const int wave = threadIdx.x >> 5;
    const int lane = threadIdx.x & 31;
    const int nstrips = (N + 15) >> 4;
    const int strip = blockIdx.x * 8 + wave;
    if (strip >= nstrips) return;            // wave-uniform exit: EXEC stays all-ones

    const int row0 = strip << 4;
    const int half = lane >> 4;              // 0: lanes 0-15, 1: lanes 16-31
    const int l16  = lane & 15;
    const bool full_strip = (row0 + 16 <= N); // wave-uniform: no EXEC churn in stores

    int mrow = row0 + l16;
    if (mrow >= N) mrow = N - 1;             // clamp loads; stores guarded below

    // ---- A operand (16x128 strip of h): bf16 hi/lo split, WMMA A layout ----
    // A 16x32, contiguous K as in packed B: lane<16 holds K{kb+0..15}; lane>=16 K{kb+16..31}
    v16bf ahi[4], alo[4];
#pragma unroll
    for (int kb = 0; kb < 4; ++kb) {
        const float* hp = h + (size_t)mrow * HID + kb * 32 + half * 16;
        float a[16];
#pragma unroll
        for (int t = 0; t < 8; ++t) { a[t] = hp[t]; a[8 + t] = hp[8 + t]; }
#pragma unroll
        for (int t = 0; t < 16; ++t) {
            __bf16 x = (__bf16)a[t];
            ahi[kb][t] = x;
            alo[kb][t] = (__bf16)(a[t] - (float)x);
        }
    }

    const v16bf* Bhi = (const v16bf*)wbhi;
    const v16bf* Blo = (const v16bf*)wblo;
    const float* Bs[3] = {bq, bk, bv};
    float*       Os[3] = {qh, kh, vh};

    for (int mat = 0; mat < 3; ++mat) {
        const float* bias  = Bs[mat];
        float*       outp  = Os[mat];
        const float  scale = (mat == 0) ? 0.25f : 1.0f;   // DH^-0.5 = 16^-0.5

        for (int c = 0; c < 8; ++c) {                     // 8 column tiles of 16
            const int j = (c << 4) + l16;                 // output feature
            const size_t tbase = (((size_t)mat * 8 + c) * 4) * 32 + lane;
            v8f acc = {0, 0, 0, 0, 0, 0, 0, 0};
#pragma unroll
            for (int kb = 0; kb < 4; ++kb) {
                v16bf bhi = Bhi[tbase + kb * 32];
                v16bf blo = Blo[tbase + kb * 32];
                acc = __builtin_amdgcn_wmma_f32_16x16x32_bf16(false, ahi[kb], false, bhi, (short)0, acc, false, false);
                acc = __builtin_amdgcn_wmma_f32_16x16x32_bf16(false, alo[kb], false, bhi, (short)0, acc, false, false);
                acc = __builtin_amdgcn_wmma_f32_16x16x32_bf16(false, ahi[kb], false, blo, (short)0, acc, false, false);
            }
            const float bj  = bias[j];
            const int   pos = ((j & 7) << 4) + (j >> 3);  // head-major: (j%8)*16 + j/8
            // C VGPR r: M = half*8 + r, N-col = l16
            float* basep = outp + (size_t)(row0 + half * 8) * HID + pos;
            if (full_strip) {                             // hot path: unconditional stores
#pragma unroll
                for (int r = 0; r < 8; ++r)
                    basep[(size_t)r * HID] = (acc[r] + bj) * scale;
            } else {                                      // cold partial strip
#pragma unroll
                for (int r = 0; r < 8; ++r) {
                    const int mm = row0 + half * 8 + r;
                    if (mm < N) outp[(size_t)mm * HID + pos] = (acc[r] + bj) * scale;
                }
            }
        }
    }
}

// ---------------------------------------------------------------------------
// CSR construction (deterministic segments): count -> scan -> scatter -> sort
// ---------------------------------------------------------------------------
__global__ void count_kernel(const int* __restrict__ src, int* __restrict__ cnt, int E)
{
    int e = blockIdx.x * blockDim.x + threadIdx.x;
    if (e < E) atomicAdd(&cnt[src[e]], 1);
}

__global__ void scan_kernel(const int* __restrict__ cnt, int* __restrict__ row_start, int n)
{
    __shared__ int sh[1024];
    const int tid = threadIdx.x;
    int offset = 0;
    if (tid == 0) row_start[0] = 0;
    for (int base = 0; base < n; base += 1024) {
        int i = base + tid;
        int x = (i < n) ? cnt[i] : 0;
        sh[tid] = x;
        __syncthreads();
        for (int d = 1; d < 1024; d <<= 1) {
            int t = (tid >= d) ? sh[tid - d] : 0;
            __syncthreads();
            sh[tid] += t;
            __syncthreads();
        }
        if (i < n) row_start[i + 1] = offset + sh[tid];
        offset += sh[1023];
        __syncthreads();
    }
}

__global__ void scatter_kernel(const int* __restrict__ src, const int* __restrict__ row_start,
                               int* __restrict__ cursor, int* __restrict__ eid, int E)
{
    int e = blockIdx.x * blockDim.x + threadIdx.x;
    if (e >= E) return;
    int s = src[e];
    int slot = atomicAdd(&cursor[s], 1);
    eid[row_start[s] + slot] = e;
}

// restore deterministic (ascending edge-id) order within each row
__global__ void sort_rows(int* __restrict__ eid, const int* __restrict__ row_start, int N)
{
    int n = blockIdx.x * blockDim.x + threadIdx.x;
    if (n >= N) return;
    int s = row_start[n], e = row_start[n + 1];
    for (int i = s + 1; i < e; ++i) {
        int key = eid[i];
        int j = i - 1;
        while (j >= s && eid[j] > key) { eid[j + 1] = eid[j]; --j; }
        eid[j + 1] = key;
    }
}

// ---------------------------------------------------------------------------
// Kernel 2: fused sparse softmax + aggregation, single pass (online softmax).
// One wave per node; lane l: head hh = l>>2, dims (l&3)*4 .. +3.
// k/v rows loaded once per edge (coalesced 512B, L2-resident: 76.8MB < 192MB).
// Deterministic: fixed sorted edge order, sequential accumulation per wave.
// ---------------------------------------------------------------------------
__global__ __launch_bounds__(256) void attn_kernel(
    const float* __restrict__ qh, const float* __restrict__ kh, const float* __restrict__ vh,
    const int* __restrict__ row_start, const int* __restrict__ eid,
    const int* __restrict__ dst, const float* __restrict__ val,
    float* __restrict__ out, int N)
{
    const int wave = threadIdx.x >> 5;
    const int lane = threadIdx.x & 31;
    const int n = blockIdx.x * 8 + wave;
    if (n >= N) return;

    const int hh = lane >> 2;
    const int d0 = (lane & 3) * 4;
    const int start = row_start[n];
    const int end   = row_start[n + 1];

    if (start == end) {                       // no outgoing edges -> zero row
#pragma unroll
        for (int t = 0; t < 4; ++t) out[(size_t)n * HID + (d0 + t) * 8 + hh] = 0.0f;
        return;
    }

    const float4 q4 = ((const float4*)(qh + (size_t)n * HID))[lane];

    float  m    = -__builtin_inff();
    float  ssum = 0.0f;
    float4 acc  = {0.0f, 0.0f, 0.0f, 0.0f};

    for (int i = start; i < end; ++i) {
        int e = eid[i];
        int u = dst[e];
        float4 k4 = ((const float4*)(kh + (size_t)u * HID))[lane];
        float d = q4.x * k4.x + q4.y * k4.y + q4.z * k4.z + q4.w * k4.w;
        d += __shfl_xor(d, 1, 32);
        d += __shfl_xor(d, 2, 32);            // all 4 lanes of a head hold the dot
        float sc = d * val[e];

        float mnew  = fmaxf(m, sc);
        float rs    = __expf(m - mnew);       // ==0 on first edge (m = -inf)
        float p     = __expf(sc - mnew);
        ssum        = ssum * rs + p;

        float4 v4 = ((const float4*)(vh + (size_t)u * HID))[lane];
        acc.x = acc.x * rs + p * v4.x;
        acc.y = acc.y * rs + p * v4.y;
        acc.z = acc.z * rs + p * v4.z;
        acc.w = acc.w * rs + p * v4.w;
        m = mnew;
    }

    const float inv = 1.0f / ssum;
    float r[4] = {acc.x * inv, acc.y * inv, acc.z * inv, acc.w * inv};
#pragma unroll
    for (int t = 0; t < 4; ++t)
        out[(size_t)n * HID + (d0 + t) * 8 + hh] = r[t];  // out[n, d*8 + hh]
}

// ---------------------------------------------------------------------------
extern "C" void kernel_launch(void* const* d_in, const int* in_sizes, int n_in,
                              void* d_out, int out_size, void* d_ws, size_t ws_size,
                              hipStream_t stream)
{
    const float* h   = (const float*)d_in[0];
    const float* val = (const float*)d_in[1];
    const float* Wq  = (const float*)d_in[2];
    const float* bq  = (const float*)d_in[3];
    const float* Wk  = (const float*)d_in[4];
    const float* bk  = (const float*)d_in[5];
    const float* Wv  = (const float*)d_in[6];
    const float* bv  = (const float*)d_in[7];
    const int*   src = (const int*)d_in[8];
    const int*   dst = (const int*)d_in[9];
    float*       out = (float*)d_out;

    const int N = in_sizes[0] / HID;
    const int E = in_sizes[8];

    // workspace carve-out (256B aligned)
    char* ws = (char*)d_ws;
    size_t off = 0;
    auto carve = [&](size_t bytes) -> char* {
        char* p = ws + off;
        off = (off + bytes + 255) & ~(size_t)255;
        return p;
    };
    float*  qh        = (float*)carve((size_t)N * HID * sizeof(float));
    float*  kh        = (float*)carve((size_t)N * HID * sizeof(float));
    float*  vh        = (float*)carve((size_t)N * HID * sizeof(float));
    __bf16* wbhi      = (__bf16*)carve((size_t)3 * 8 * 4 * 32 * 16 * sizeof(__bf16));
    __bf16* wblo      = (__bf16*)carve((size_t)3 * 8 * 4 * 32 * 16 * sizeof(__bf16));
    int*    cnt       = (int*)carve((size_t)N * sizeof(int));
    int*    cursor    = (int*)carve((size_t)N * sizeof(int));
    int*    row_start = (int*)carve((size_t)(N + 1) * sizeof(int));
    int*    eid       = (int*)carve((size_t)E * sizeof(int));

    (void)hipMemsetAsync(cnt,    0, (size_t)N * sizeof(int), stream);
    (void)hipMemsetAsync(cursor, 0, (size_t)N * sizeof(int), stream);

    pack_w_kernel<<<12, 256, 0, stream>>>(Wq, Wk, Wv, wbhi, wblo);

    const int strips = (N + 15) / 16;
    qkv_proj_wmma<<<(strips + 7) / 8, 256, 0, stream>>>(h, wbhi, wblo, bq, bk, bv,
                                                        qh, kh, vh, N);
    count_kernel  <<<(E + 255) / 256, 256, 0, stream>>>(src, cnt, E);
    scan_kernel   <<<1, 1024, 0, stream>>>(cnt, row_start, N);
    scatter_kernel<<<(E + 255) / 256, 256, 0, stream>>>(src, row_start, cursor, eid, E);
    sort_rows     <<<(N + 255) / 256, 256, 0, stream>>>(eid, row_start, N);
    attn_kernel   <<<(N + 7) / 8, 256, 0, stream>>>(qh, kh, vh, row_start, eid, dst, val, out, N);
}
